// ObjectRepresentation_6897717477444
// MI455X (gfx1250) — matrix operand order repaired and checked by reading, not compiled, not run on hardware
//
#include <hip/hip_runtime.h>
#include <stdint.h>

// ---------------- problem dims ----------------
#define NC_ 4
#define BS_ 16
#define H_  64
#define W_  64
#define OC_ 12
#define NT_ 32

static constexpr int NSTATE = BS_ * H_ * W_ * OC_;   // 786432 floats per state slab
static constexpr int NCLASS = NC_ * NSTATE;          // 3145728 (class bank)

// gen cell: Cin = NC*OC + OC = 60, K = 9*60 = 540, padded to 17*32 = 544
// class cell: Cin = OC*3 = 36,     K = 9*36 = 324, padded to 11*32 = 352
static constexpr int KS_G = 17, KTOT_G = 540, CIN_G = 60;
static constexpr int KS_C = 11, KTOT_C = 324, CIN_C = 36;
static constexpr int NPACK_G = KS_G * 3 * 32 * 16;   // 26112 halfs
static constexpr int NPACK_C = KS_C * 3 * 32 * 16;   // 16896 halfs

typedef __attribute__((ext_vector_type(16))) _Float16 v16h;
typedef __attribute__((ext_vector_type(8)))  _Float16 v8h;
typedef __attribute__((ext_vector_type(8)))  float    v8f;

// ---------------- init ----------------
__global__ void zero_init(float* a, size_t na, float* b, size_t nb) {
    size_t i = (size_t)blockIdx.x * blockDim.x + threadIdx.x;
    size_t stride = (size_t)gridDim.x * blockDim.x;
    for (size_t k = i; k < na; k += stride) a[k] = 0.0f;
    for (size_t k = i; k < nb; k += stride) b[k] = 0.0f;
}

// ---------------- weight pack (f32 HWIO -> f16 WMMA B-fragment layout) ----------------
// Packed element ((ks*3+nt)*32 + lane)*16 + j  holds  W[K][N] with
//   K = ks*32 + (lane>>4)*16 + j,  N = nt*16 + (lane&15)
__global__ void pack_weights(const float* __restrict__ Wxg, const float* __restrict__ Whg,
                             const float* __restrict__ Wxc, const float* __restrict__ Whc,
                             _Float16* __restrict__ wpg, _Float16* __restrict__ wpc) {
    int idx = blockIdx.x * 256 + threadIdx.x;
    if (idx >= NPACK_G + NPACK_C) return;
    bool gen = idx < NPACK_G;
    int local = gen ? idx : idx - NPACK_G;
    int j    = local & 15;
    int lane = (local >> 4) & 31;
    int nt   = (local >> 9) % 3;
    int ks   = local / 1536;
    int hi = lane >> 4, lm = lane & 15;
    int n  = nt * 16 + lm;
    int K  = ks * 32 + hi * 16 + j;
    float v = 0.0f;
    if (gen) {
        if (K < KTOT_G) {
            int tap = K / CIN_G, c = K - tap * CIN_G;   // tap = ky*3+kx
            v = (c < 48) ? Wxg[(tap * 48 + c) * 48 + n]
                         : Whg[(tap * 12 + (c - 48)) * 48 + n];
        }
        wpg[local] = (_Float16)v;
    } else {
        if (K < KTOT_C) {
            int tap = K / CIN_C, c = K - tap * CIN_C;
            v = (c < 24) ? Wxc[(tap * 24 + c) * 48 + n]
                         : Whc[(tap * 12 + (c - 24)) * 48 + n];
        }
        wpc[local] = (_Float16)v;
    }
}

// ---------------- fused ConvLSTM cell: implicit-GEMM conv via WMMA + gates ----------------
// One block per (batch, image row): 128 threads = 4 waves, each wave owns a
// 16-pixel x 48-channel z tile. K-mapping: k = (ky*3+kx)*CIN + c  (shared by
// im2col builder and B-pack, zero padded to KSTEPS*32).
template <bool IS_GEN, int CIN, int KSTEPS, int KTOT>
__launch_bounds__(128)
__global__ void convlstm_cell(const float* __restrict__ classH,   // gen: x_gen bank; class: recurrent bank
                              const float* __restrict__ genH_in,  // gen: h(t-1);   class: fresh gen h
                              const float* __restrict__ frames,   // class only: x (BS,NT,H,W,OC)
                              const _Float16* __restrict__ wpack,
                              const float* __restrict__ bias,
                              const float* __restrict__ classC,   // class: old c bank
                              float* __restrict__ genC,           // gen: c state (in-place)
                              float* __restrict__ tmpC,           // class: new c out
                              float* __restrict__ h_out,
                              const int* __restrict__ class_ids, int t) {
    constexpr int KPAD = KSTEPS * 32;
    constexpr int HALO = 3 * 66 * CIN;                         // halfs
    constexpr int HALO_B = (HALO * 2 + 15) & ~15;              // bytes, 16-aligned

    extern __shared__ char smem[];
    _Float16* halo = (_Float16*)smem;                          // stage-1 halo patch
    _Float16* apix = (_Float16*)(smem + HALO_B);               // im2col: [64][KPAD]

    const int tid = threadIdx.x;
    const int blk = blockIdx.x;
    const int b = blk >> 6;          // / H
    const int y = blk & 63;
    int cid = 0;
    if constexpr (!IS_GEN) cid = class_ids[t];

    // ---- stage 1a: stage halo patch [ry 0..2][lx 0..65][c 0..CIN) as f16, zero pad ----
    for (int i = tid; i < HALO; i += 128) {
        int c = i % CIN;
        int rest = i / CIN;
        int lx = rest % 66;
        int ry = rest / 66;
        int yy = y + ry - 1;
        int xx = lx - 1;
        float v = 0.0f;
        if ((unsigned)yy < (unsigned)H_ && (unsigned)xx < (unsigned)W_) {
            int sp = (b * H_ + yy) * W_ + xx;
            if constexpr (IS_GEN) {
                // x_gen = raw reshape of class_h bank to (BS,H,W,48)
                v = (c < 48) ? classH[sp * 48 + c] : genH_in[sp * OC_ + (c - 48)];
            } else {
                if (c < 12)      v = frames[(size_t)(((b * NT_ + t) * H_ + yy) * W_ + xx) * OC_ + c];
                else if (c < 24) v = genH_in[sp * OC_ + (c - 12)];
                else             v = classH[(size_t)cid * NSTATE + sp * OC_ + (c - 24)];
            }
        }
        halo[i] = (_Float16)v;
    }
    // ---- stage 1b: zero the K-padding tail of the im2col matrix ----
    constexpr int TAIL = KPAD - KTOT;
    for (int i = tid; i < 64 * TAIL; i += 128) {
        int p = i / TAIL;
        int kk = i - p * TAIL;
        apix[p * KPAD + KTOT + kk] = (_Float16)0.0f;
    }
    __syncthreads();

    // ---- stage 1c: im2col expansion halo -> apix via wide LDS b64 copies ----
    // run (x, tap): apix[x][tap*CIN .. +CIN) = halo[(ky*66 + x+kx)*CIN .. +CIN)
    {
        const uint64_t* h64 = (const uint64_t*)halo;
        uint64_t* a64 = (uint64_t*)apix;
        constexpr int RUN64 = CIN / 4;            // b64 chunks per run (15 gen, 9 class)
        for (int r = tid; r < 64 * 9; r += 128) {
            int x = r / 9;
            int tap = r - x * 9;
            int ky = tap / 3;
            int kx = tap - ky * 3;
            int src = ((ky * 66 + x + kx) * CIN) >> 2;   // u64 units
            int dst = (x * KPAD + tap * CIN) >> 2;
#pragma unroll
            for (int q = 0; q < RUN64; ++q) a64[dst + q] = h64[src + q];
        }
    }
    __syncthreads();

    // ---- stage 2: WMMA K-loop: 2x ds_load_b128 (A) + 3x 32B global (B) + 3 WMMA ----
    const int lane = tid & 31;
    const int mt = tid >> 5;          // wave id: pixel subtile
    const int hi = lane >> 4;
    const int lm = lane & 15;
    const int x0 = mt * 16 + lm;      // pixel this lane's A row maps to

    v8f acc0 = {}, acc1 = {}, acc2 = {};
    const v16h* wp = (const v16h*)wpack;
    const _Float16* ap = apix + x0 * KPAD + hi * 16;
    for (int ks = 0; ks < KSTEPS; ++ks) {
        // A 16x32 f16 layout: element j is K = ks*32+hi*16+j (contiguous,
        // matching the B pack)
        v8h alo = *(const v8h*)(ap + ks * 32);
        v8h ahi = *(const v8h*)(ap + ks * 32 + 8);
        v16h a = __builtin_shufflevector(alo, ahi, 0, 1, 2, 3, 4, 5, 6, 7,
                                         8, 9, 10, 11, 12, 13, 14, 15);
        v16h b0 = wp[(ks * 3 + 0) * 32 + lane];
        v16h b1 = wp[(ks * 3 + 1) * 32 + lane];
        v16h b2 = wp[(ks * 3 + 2) * 32 + lane];
        acc0 = __builtin_amdgcn_wmma_f32_16x16x32_f16(false, a, false, b0, (short)0, acc0, false, false);
        acc1 = __builtin_amdgcn_wmma_f32_16x16x32_f16(false, a, false, b1, (short)0, acc1, false, false);
        acc2 = __builtin_amdgcn_wmma_f32_16x16x32_f16(false, a, false, b2, (short)0, acc2, false, false);
    }
    __syncthreads();   // halo/apix dead; reuse front of smem for z

    // ---- stage 3: de-stripe C layout (VGPR r -> M=r+8*hi, lane -> N) into pixel-major z ----
    float* zb = (float*)smem;
#pragma unroll
    for (int r = 0; r < 8; ++r) {
        int p = mt * 16 + r + 8 * hi;
        zb[p * 48 + 0 + lm]  = acc0[r];
        zb[p * 48 + 16 + lm] = acc1[r];
        zb[p * 48 + 32 + lm] = acc2[r];
    }
    __syncthreads();

    // ---- stage 4: gates (jax hard_sigmoid = clamp(x/6+0.5,0,1)) + state update ----
    for (int idx = tid; idx < 64 * OC_; idx += 128) {
        int p = idx / OC_;
        int c = idx - p * OC_;
        float zi = zb[p * 48 + c]      + bias[c];
        float zf = zb[p * 48 + 12 + c] + bias[12 + c];
        float zg = zb[p * 48 + 24 + c] + bias[24 + c];
        float zo = zb[p * 48 + 36 + c] + bias[36 + c];
        float ig = fminf(fmaxf(zi * (1.0f / 6.0f) + 0.5f, 0.0f), 1.0f);
        float fg = fminf(fmaxf(zf * (1.0f / 6.0f) + 0.5f, 0.0f), 1.0f);
        float og = fminf(fmaxf(zo * (1.0f / 6.0f) + 0.5f, 0.0f), 1.0f);
        float gg = tanhf(zg);
        int li = ((b * H_ + y) * W_ + p) * OC_ + c;
        if constexpr (IS_GEN) {
            float cold = genC[li];
            float cnew = fg * cold + ig * gg;
            genC[li] = cnew;
            h_out[li] = og * tanhf(cnew);
        } else {
            float cold = classC[(size_t)cid * NSTATE + li];
            float cnew = fg * cold + ig * gg;
            tmpC[li] = cnew;
            h_out[li] = og * tanhf(cnew);
        }
    }
}

// ---------------- commit: scatter tmp into class bank slot cid, accumulate output ----------------
__global__ void commit_step(float* __restrict__ classH, float* __restrict__ classC,
                            const float* __restrict__ tmpH, const float* __restrict__ tmpC,
                            float* __restrict__ out, const int* __restrict__ cids, int t) {
    int i = blockIdx.x * 256 + threadIdx.x;   // grid sized exactly NSTATE/256
    int cid = cids[t];
    size_t si = (size_t)cid * NSTATE + i;
    float h = tmpH[i];
    classH[si] = h;
    classC[si] = tmpC[i];
    out[i] += h;
}

// ---------------- host ----------------
extern "C" void kernel_launch(void* const* d_in, const int* in_sizes, int n_in,
                              void* d_out, int out_size, void* d_ws, size_t ws_size,
                              hipStream_t stream) {
    (void)in_sizes; (void)n_in; (void)out_size; (void)ws_size;
    const float* x    = (const float*)d_in[0];
    const int*   cids = (const int*)d_in[1];
    const float* Wxg  = (const float*)d_in[2];
    const float* Whg  = (const float*)d_in[3];
    const float* bg   = (const float*)d_in[4];
    const float* Wxc  = (const float*)d_in[5];
    const float* Whc  = (const float*)d_in[6];
    const float* bc   = (const float*)d_in[7];
    float* out = (float*)d_out;

    float* ws     = (float*)d_ws;
    float* classH = ws;
    float* classC = classH + NCLASS;
    float* genH0  = classC + NCLASS;
    float* genH1  = genH0 + NSTATE;
    float* genC   = genH1 + NSTATE;
    float* tmpH   = genC + NSTATE;
    float* tmpC   = tmpH + NSTATE;
    _Float16* wpg = (_Float16*)(tmpC + NSTATE);
    _Float16* wpc = wpg + NPACK_G;

    // dynamic LDS: halo (16B aligned) + im2col matrix
    constexpr int SB_G = ((3 * 66 * CIN_G * 2 + 15) & ~15) + 64 * (KS_G * 32) * 2;  // 93392 B
    constexpr int SB_C = ((3 * 66 * CIN_C * 2 + 15) & ~15) + 64 * (KS_C * 32) * 2;  // 59312 B

    // zero state banks (classH, classC, genH0, genH1, genC) and output accumulator
    zero_init<<<2048, 256, 0, stream>>>(classH, (size_t)(2 * NCLASS + 3 * NSTATE),
                                        out, (size_t)NSTATE);
    pack_weights<<<(NPACK_G + NPACK_C + 255) / 256, 256, 0, stream>>>(Wxg, Whg, Wxc, Whc, wpg, wpc);

    for (int t = 0; t < NT_; ++t) {
        float* hprev = (t & 1) ? genH1 : genH0;
        float* hnew  = (t & 1) ? genH0 : genH1;
        // general cell: reads full class_h bank (raw reshape) + gen_h(t-1)
        convlstm_cell<true, CIN_G, KS_G, KTOT_G><<<BS_ * H_, 128, SB_G, stream>>>(
            classH, hprev, nullptr, wpg, bg, nullptr, genC, nullptr, hnew, nullptr, t);
        // class cell: reads frame t + fresh gen_h + class_h[cid] / class_c[cid]
        convlstm_cell<false, CIN_C, KS_C, KTOT_C><<<BS_ * H_, 128, SB_C, stream>>>(
            classH, hnew, x, wpc, bc, classC, nullptr, tmpC, tmpH, cids, t);
        // scatter into class bank + accumulate output sum
        commit_step<<<NSTATE / 256, 256, 0, stream>>>(classH, classC, tmpH, tmpC, out, cids, t);
    }
}
